// LlamaAttention_3307124817936
// MI455X (gfx1250) — compile-verified, matrix operand order checked by reading, not run on hardware
//
#include <hip/hip_runtime.h>
#include <cstdint>
#include <cstddef>

typedef __attribute__((ext_vector_type(16))) __bf16 v16bf;
typedef __attribute__((ext_vector_type(8)))  float  v8f;

constexpr int kB = 2, kL = 2048, kD = 2048, kH = 32, kKVH = 8, kHD = 64;
constexpr int kM   = kB * kL;        // 4096 GEMM rows
constexpr int kKVD = kKVH * kHD;     // 512

// ---------------------------------------------------------------------------
// helpers
// ---------------------------------------------------------------------------
__device__ __forceinline__ __bf16 to_bf16(float v)  { return (__bf16)v; }
__device__ __forceinline__ __bf16 to_bf16(__bf16 v) { return v; }

__device__ __forceinline__ v8f v8f_zero() {
  v8f z = {0.f, 0.f, 0.f, 0.f, 0.f, 0.f, 0.f, 0.f};
  return z;
}

// CDNA5 async copy: per-lane 16B global -> LDS, tracked by ASYNCcnt.
// (ISA 15.18.3 GLOBAL_LOAD_ASYNC_TO_LDS_B128; inline asm is portable across
//  ROCm7.2/amdgpu-toolchain builtin arity differences.)
// ldsOff = low 32 bits of generic LDS pointer (= byte offset in group segment)
__device__ __forceinline__ void async_copy_b128(void* ldsPtr, const void* gPtr) {
  uint32_t ldsOff = (uint32_t)(uintptr_t)ldsPtr;
  uint64_t gaddr  = (uint64_t)(uintptr_t)gPtr;
  asm volatile("global_load_async_to_lds_b128 %0, %1, off"
               :: "v"(ldsOff), "v"(gaddr) : "memory");
}
__device__ __forceinline__ void wait_async0() {
  asm volatile("s_wait_asynccnt 0x0" ::: "memory");
}

// Load a 16x32 bf16 fragment in the CDNA5 WMMA 16-bit A/B register layout:
//   lanes 0-15  : row = lane,     K = {0..7, 16..23}
//   lanes 16-31 : row = lane-16,  K = {8..15, 24..31}
// `base` points at (row0, k0) of a row-major tile with row stride `stride`.
__device__ __forceinline__ v16bf load_frag(const __bf16* base, int stride) {
  const int lane = threadIdx.x & 31;
  const int r    = lane & 15;
  const int h8   = (lane >> 4) << 3;
  const __bf16* row = base + r * stride;
  v16bf f;
#pragma unroll
  for (int j = 0; j < 8; ++j) {
    const int k = ((j < 4) ? (2 * j) : (2 * j + 8)) + h8;
    f[2 * j]     = row[k];
    f[2 * j + 1] = row[k + 1];
  }
  return f;
}

// ---------------------------------------------------------------------------
// GEMM core: C[rowBase.., colOut..] = bf16(A) @ bf16(W)
// A: M x K row-major (lda), element type float (VGPR cvt path) or
//    __bf16 (async LDS-copy path).  W: K x ldw row-major fp32.
// Block: 256 threads (8 waves), tile 128(M) x 128(N), K-step 32.
// Wave (2x4 grid) computes a 64x32 tile = 4x2 WMMA fragments.
// ---------------------------------------------------------------------------
template <typename TAelem>
__device__ void gemm_core(const TAelem* __restrict__ A, int lda,
                          const float* __restrict__ W, int ldw,
                          float* __restrict__ Cout, int ldc,
                          int rowBase, int colW, int colOut, int K) {
  __shared__ __bf16 Xs[128][40];   // [m][k]  pad 32->40: rows 80B, 16B-aligned
  __shared__ __bf16 Ws[128][40];   // [n][k]  (W staged transposed)

  const int tid  = threadIdx.x;
  const int wave = tid >> 5;
  const int wm   = (wave >> 2) * 64;
  const int wn   = (wave & 3) * 32;

  v8f acc[4][2];
#pragma unroll
  for (int i = 0; i < 4; ++i)
#pragma unroll
    for (int j = 0; j < 2; ++j) acc[i][j] = v8f_zero();

  for (int k0 = 0; k0 < K; k0 += 32) {
    // ---- stage A tile: 128 x 32
    if constexpr (sizeof(TAelem) == 2) {
      // bf16 input: pure byte move -> async copy engine (2 x 256 lanes x 16B)
      const int rowA = tid >> 2;          // 0..63
      const int sg   = tid & 3;           // 4 x 8 bf16 = 32 per row
      const TAelem* src0 = A + (size_t)(rowBase + rowA) * lda + k0 + sg * 8;
      const TAelem* src1 = src0 + (size_t)64 * lda;
      if (k0 + 32 < K) __builtin_prefetch(src0 + 32, 0, 1);
      async_copy_b128(&Xs[rowA][sg * 8], src0);
      async_copy_b128(&Xs[64 + rowA][sg * 8], src1);
    } else {
      // fp32 input: load + v_cvt_pk_bf16 in VGPRs
      const int rowL = tid >> 1;
      const int half = tid & 1;
      const TAelem* src = A + (size_t)(rowBase + rowL) * lda + k0 + half * 16;
      if (k0 + 32 < K) __builtin_prefetch(src + 32, 0, 1);
#pragma unroll
      for (int j = 0; j < 16; ++j) Xs[rowL][half * 16 + j] = to_bf16(src[j]);
    }
    // ---- stage W tile transposed: Ws[n][k] = W[k0+k][colW+n]  (fp32 -> bf16)
    {
      const int kk = tid >> 3;
      const int ng = tid & 7;
      const float* src = W + (size_t)(k0 + kk) * ldw + colW + ng * 16;
      if (k0 + 32 < K) __builtin_prefetch(src + (size_t)32 * ldw, 0, 1);
#pragma unroll
      for (int j = 0; j < 16; ++j) Ws[ng * 16 + j][kk] = to_bf16(src[j]);
    }
    if constexpr (sizeof(TAelem) == 2) wait_async0();
    __syncthreads();

    v16bf af[4], bfr[2];
#pragma unroll
    for (int i = 0; i < 4; ++i) af[i]  = load_frag(&Xs[wm + i * 16][0], 40);
#pragma unroll
    for (int j = 0; j < 2; ++j) bfr[j] = load_frag(&Ws[wn + j * 16][0], 40);
#pragma unroll
    for (int i = 0; i < 4; ++i)
#pragma unroll
      for (int j = 0; j < 2; ++j)
        acc[i][j] = __builtin_amdgcn_wmma_f32_16x16x32_bf16(
            false, af[i], false, bfr[j], (short)0, acc[i][j], false, false);
    __syncthreads();
  }

  // C fragment layout: lanes 0-15 -> N=lane, M=r ; lanes 16-31 -> N=lane-16, M=8+r
  const int lane = tid & 31;
  const int nIdx = lane & 15;
  const int mAdd = (lane >> 4) << 3;
#pragma unroll
  for (int i = 0; i < 4; ++i)
#pragma unroll
    for (int j = 0; j < 2; ++j) {
      float* dst = Cout + (size_t)(rowBase + wm + i * 16 + mAdd) * ldc +
                   colOut + wn + j * 16 + nIdx;
#pragma unroll
      for (int r = 0; r < 8; ++r) dst[(size_t)r * ldc] = acc[i][j][r];
    }
}

// ---------------------------------------------------------------------------
// Kernel 1: fused QKV projection.  N range 0..3071 = Q(2048) | K(512) | V(512)
// ---------------------------------------------------------------------------
__global__ __launch_bounds__(256) void qkv_gemm_kernel(
    const float* __restrict__ x,
    const float* __restrict__ Wq, const float* __restrict__ Wk,
    const float* __restrict__ Wv,
    float* __restrict__ qtmp, float* __restrict__ ktmp,
    float* __restrict__ vtmp) {
  const int c0 = blockIdx.x * 128;        // 24 tiles
  const int rowBase = blockIdx.y * 128;   // 32 tiles

  const float* W; float* out; int ldw, colW, ldc;
  if (c0 < kD)            { W = Wq; out = qtmp; ldw = kD;   colW = c0;          ldc = kD;   }
  else if (c0 < kD + kKVD){ W = Wk; out = ktmp; ldw = kKVD; colW = c0 - kD;     ldc = kKVD; }
  else                    { W = Wv; out = vtmp; ldw = kKVD; colW = c0 - kD - kKVD; ldc = kKVD; }

  gemm_core<float>(x, kD, W, ldw, out, ldc, rowBase, colW, colW, kD);
}

// ---------------------------------------------------------------------------
// Kernel 2: RoPE + head-major relayout (+ V transpose), fp32 -> bf16
//   qh: (B,H,L,HD)  kh: (B,KVH,L,HD)  vt: (B,KVH,HD,L)
// ---------------------------------------------------------------------------
__global__ __launch_bounds__(256) void rope_kernel(
    const float* __restrict__ qtmp, const float* __restrict__ ktmp,
    const float* __restrict__ vtmp,
    const float* __restrict__ cosT, const float* __restrict__ sinT,
    __bf16* __restrict__ qh, __bf16* __restrict__ kh,
    __bf16* __restrict__ vt) {
  const int QN = kB * kH * kL * kHD;     // 8388608
  const int KN = kB * kKVH * kL * kHD;   // 2097152
  const int idx = blockIdx.x * blockDim.x + threadIdx.x;

  if (idx < QN) {                                   // Q with RoPE
    const int d = idx & 63, l = (idx >> 6) & 2047;
    const int h = (idx >> 17) & 31, b = idx >> 22;
    const size_t src = (size_t)(b * kL + l) * kD + h * kHD;
    const float v  = qtmp[src + d];
    const float vr = (d < 32) ? -qtmp[src + d + 32] : qtmp[src + d - 32];
    qh[idx] = (__bf16)(v * cosT[l * kHD + d] + vr * sinT[l * kHD + d]);
  } else if (idx < QN + KN) {                       // K with RoPE
    const int j = idx - QN;
    const int d = j & 63, l = (j >> 6) & 2047;
    const int h = (j >> 17) & 7, b = j >> 20;
    const size_t src = (size_t)(b * kL + l) * kKVD + h * kHD;
    const float v  = ktmp[src + d];
    const float vr = (d < 32) ? -ktmp[src + d + 32] : ktmp[src + d - 32];
    kh[j] = (__bf16)(v * cosT[l * kHD + d] + vr * sinT[l * kHD + d]);
  } else if (idx < QN + 2 * KN) {                   // V transpose
    const int j = idx - QN - KN;                    // vt linear index
    const int l = j & 2047, d = (j >> 11) & 63;
    const int h = (j >> 17) & 7, b = j >> 20;
    vt[j] = (__bf16)vtmp[(size_t)(b * kL + l) * kKVD + h * kHD + d];
  }
}

// ---------------------------------------------------------------------------
// Kernel 3: causal GQA flash attention.
// Block = 128 threads (4 waves); each wave owns 16 query rows; block owns 64.
// Computes S^T = K*Q^T so the softmaxed P lands directly in the A-fragment
// layout for P*V (no LDS transpose). K/V tiles staged with the CDNA5 async
// copy engine (global_load_async_to_lds_b128 / ASYNCcnt).
// ---------------------------------------------------------------------------
__global__ __launch_bounds__(128) void attn_kernel(
    const __bf16* __restrict__ qh, const __bf16* __restrict__ kh,
    const __bf16* __restrict__ vt, __bf16* __restrict__ attnOut) {
  __shared__ __bf16 Ks[32][72];   // [key][d]  rows 144B (16B-aligned), pad 8
  __shared__ __bf16 Vs[64][40];   // [d][key]  rows  80B (16B-aligned), pad 8

  const int bh   = blockIdx.y;            // 0..63
  const int b    = bh >> 5, h = bh & 31;
  const int kvh  = h >> 2;                // N_REP = 4
  const int qBlk = blockIdx.x;            // 0..31
  const int wave = threadIdx.x >> 5;
  const int lane = threadIdx.x & 31;
  const int q0   = qBlk * 64 + wave * 16;

  const __bf16* Qp = qh + ((size_t)(b * kH + h) * kL + q0) * kHD;
  const __bf16* Kp = kh + (size_t)(b * kKVH + kvh) * kL * kHD;
  const __bf16* Vp = vt + (size_t)(b * kKVH + kvh) * kHD * kL;   // [d][l]

  // Q as B-fragment (N=q rows, K=HD), two 32-wide K chunks, held all loop long
  v16bf bq0 = load_frag(Qp, kHD);
  v16bf bq1 = load_frag(Qp + 32, kHD);

  v8f o[4];
#pragma unroll
  for (int dt = 0; dt < 4; ++dt) o[dt] = v8f_zero();
  float mi = -1e30f, li = 0.f;

  const int nk = qBlk * 64 + 64;          // causal bound (uniform per block)
  for (int kt = 0; kt < nk; kt += 32) {
    {  // async-stage K tile (32 keys x 64) and V^T tile (64 d x 32): 16B/lane
      const int t   = threadIdx.x;
      const int key = t >> 3;             // 0..15 per copy
      const int seg = t & 7;              // 8 x (8 bf16) = 64
      async_copy_b128(&Ks[key][seg * 8],
                      Kp + (size_t)(kt + key) * kHD + seg * 8);
      async_copy_b128(&Ks[16 + key][seg * 8],
                      Kp + (size_t)(kt + 16 + key) * kHD + seg * 8);
      const int d2 = t >> 2;              // 0..31 per copy
      const int sg = t & 3;               // 4 x (8 bf16) = 32
      async_copy_b128(&Vs[d2][sg * 8],
                      Vp + (size_t)d2 * kL + kt + sg * 8);
      async_copy_b128(&Vs[32 + d2][sg * 8],
                      Vp + (size_t)(32 + d2) * kL + kt + sg * 8);
      wait_async0();
    }
    __syncthreads();

    // S^T tile (32 keys x 16 q) = two C fragments; A = K rows, B = Q rows
    v8f s[2];
#pragma unroll
    for (int kk = 0; kk < 2; ++kk) {
      v16bf ak0 = load_frag(&Ks[kk * 16][0], 72);
      v16bf ak1 = load_frag(&Ks[kk * 16][32], 72);
      v8f c = v8f_zero();
      c = __builtin_amdgcn_wmma_f32_16x16x32_bf16(false, ak0, false, bq0,
                                                  (short)0, c, false, false);
      c = __builtin_amdgcn_wmma_f32_16x16x32_bf16(false, ak1, false, bq1,
                                                  (short)0, c, false, false);
      s[kk] = c;
    }

    // per-lane: lane holds q = q0 + (lane&15); its 16 key values are
    // frag0/frag1 rows with key = kt + kk*16 + r + 8*(lane>=16)
    const int q       = q0 + (lane & 15);
    const int keyBase = kt + ((lane >> 4) << 3);
    float mmax = -1e30f;
#pragma unroll
    for (int kk = 0; kk < 2; ++kk)
#pragma unroll
      for (int r = 0; r < 8; ++r) {
        const int key = keyBase + kk * 16 + r;
        float val = s[kk][r] * 0.125f;              // 1/sqrt(64)
        val = (key <= q) ? val : -1e30f;            // causal mask
        s[kk][r] = val;
        mmax = fmaxf(mmax, val);
      }
    mmax = fmaxf(mmax, __shfl_xor(mmax, 16, 32));
    const float mnew = fmaxf(mi, mmax);
    const float sc   = __expf(mi - mnew);
    float psum = 0.f;
#pragma unroll
    for (int kk = 0; kk < 2; ++kk)
#pragma unroll
      for (int r = 0; r < 8; ++r) {
        const float p = __expf(s[kk][r] - mnew);
        s[kk][r] = p;
        psum += p;
      }
    psum += __shfl_xor(psum, 16, 32);
    li = li * sc + psum;
    mi = mnew;

    // rescale O: O-frag row r maps to q-row (r + 8*(lane>=16)); gather factor
    const int srcBase = (lane >> 4) << 3;
#pragma unroll
    for (int r = 0; r < 8; ++r) {
      const float fr = __shfl(sc, srcBase + r, 32);
#pragma unroll
      for (int dt = 0; dt < 4; ++dt) o[dt][r] *= fr;
    }

    // P in S^T-fragment layout IS the A-fragment layout for P(16q x 32key)
    v16bf pa;
#pragma unroll
    for (int r = 0; r < 8; ++r) {
      pa[r]     = (__bf16)s[0][r];
      pa[8 + r] = (__bf16)s[1][r];
    }
#pragma unroll
    for (int dt = 0; dt < 4; ++dt) {
      v16bf bv = load_frag(&Vs[dt * 16][0], 40);    // B = V^T rows (N=d,K=key)
      o[dt] = __builtin_amdgcn_wmma_f32_16x16x32_bf16(
          false, pa, false, bv, (short)0, o[dt], false, false);
    }
    __syncthreads();
  }

  // normalize and write to (B, L, H, HD) bf16
  const int srcBase = (lane >> 4) << 3;
  float linv[8];
#pragma unroll
  for (int r = 0; r < 8; ++r) linv[r] = 1.f / __shfl(li, srcBase + r, 32);
  const int dLane = lane & 15;
#pragma unroll
  for (int dt = 0; dt < 4; ++dt)
#pragma unroll
    for (int r = 0; r < 8; ++r) {
      const int qq = q0 + srcBase + r;
      const int dd = dt * 16 + dLane;
      attnOut[((size_t)(b * kL + qq) * kH + h) * kHD + dd] =
          (__bf16)(o[dt][r] * linv[r]);
    }
}

// ---------------------------------------------------------------------------
// Kernel 4: output projection  out = attn(bf16, 4096x2048) @ Wo(2048x2048)
// (bf16 A operand -> async LDS-copy staging path inside gemm_core)
// ---------------------------------------------------------------------------
__global__ __launch_bounds__(256) void out_gemm_kernel(
    const __bf16* __restrict__ attn, const float* __restrict__ Wo,
    float* __restrict__ out) {
  gemm_core<__bf16>(attn, kD, Wo, kD, out, kD, blockIdx.y * 128,
                    blockIdx.x * 128, blockIdx.x * 128, kD);
}

// ---------------------------------------------------------------------------
// launch
// ---------------------------------------------------------------------------
extern "C" void kernel_launch(void* const* d_in, const int* in_sizes, int n_in,
                              void* d_out, int out_size, void* d_ws,
                              size_t ws_size, hipStream_t stream) {
  const float* x    = (const float*)d_in[0];
  const float* cosT = (const float*)d_in[1];
  const float* sinT = (const float*)d_in[2];
  // d_in[3] = causal mask (bool) — computed analytically instead
  const float* Wq   = (const float*)d_in[4];
  const float* Wk   = (const float*)d_in[5];
  const float* Wv   = (const float*)d_in[6];
  const float* Wo   = (const float*)d_in[7];
  float* out = (float*)d_out;

  char* w = (char*)d_ws;
  auto take = [&](size_t bytes) -> char* {
    char* p = w;
    w += (bytes + 255) & ~(size_t)255;
    return p;
  };
  float*  qtmp = (float*)take((size_t)kM * kD * 4);               // 32 MB
  float*  ktmp = (float*)take((size_t)kM * kKVD * 4);             //  8 MB
  float*  vtmp = (float*)take((size_t)kM * kKVD * 4);             //  8 MB
  __bf16* qh   = (__bf16*)take((size_t)kB * kH * kL * kHD * 2);   // 16 MB
  __bf16* kh   = (__bf16*)take((size_t)kB * kKVH * kL * kHD * 2); //  4 MB
  __bf16* vt   = (__bf16*)take((size_t)kB * kKVH * kHD * kL * 2); //  4 MB
  __bf16* attn = (__bf16*)take((size_t)kM * kD * 2);              // 16 MB

  qkv_gemm_kernel<<<dim3(24, 32), 256, 0, stream>>>(x, Wq, Wk, Wv, qtmp, ktmp,
                                                    vtmp);
  const int ropeTotal = kB * kH * kL * kHD + 2 * kB * kKVH * kL * kHD;
  rope_kernel<<<dim3((ropeTotal + 255) / 256), 256, 0, stream>>>(
      qtmp, ktmp, vtmp, cosT, sinT, qh, kh, vt);
  attn_kernel<<<dim3(kL / 64, kB * kH), 128, 0, stream>>>(qh, kh, vt, attn);
  out_gemm_kernel<<<dim3(kD / 128, kM / 128), 256, 0, stream>>>(attn, Wo, out);
}
